// EnhancedTransformerModel_57114475102462
// MI455X (gfx1250) — matrix operand order repaired and checked by reading, not compiled, not run on hardware
//
#include <hip/hip_runtime.h>
#include <hip/hip_bf16.h>
#include <math.h>
#include <stdint.h>

// ---------------------------------------------------------------------------
// MI455X / gfx1250: wave32, WMMA 16x16x32 bf16 -> f32 accumulate.
// - WMMA fragments assembled with ds_load_b128 from K-contiguous LDS rows.
// - Generic GEMM streams the weight tile with GLOBAL_LOAD_ASYNC_TO_LDS_B128
//   (ASYNCcnt-tracked DMA, no VGPR staging) overlapped with the VGPR-staged
//   bf16 conversion of the activation tile.
// ---------------------------------------------------------------------------
typedef __bf16 bf16_t;
typedef __attribute__((ext_vector_type(16))) __bf16 v16bf;
typedef __attribute__((ext_vector_type(8)))  __bf16 v8bf;
typedef __attribute__((ext_vector_type(4)))  __bf16 v4bf;
typedef __attribute__((ext_vector_type(8)))  float  v8f;

#define B_  4
#define S_  256
#define E_  512
#define H_  8
#define HD_ 64
#define F_  2048
#define L_  4
#define V_  32000
#define M_  (B_ * S_)
#define SQRT_E 22.627416997969522f
#define LDK 72   // bf16 LDS row stride for a 64-wide K tile (144B, 16B-aligned)
#define LDKF 68  // f32 LDS row stride for a 64-wide K tile (272B, 16B-aligned)

union FragU { v16bf v; v8bf h[2]; };

// A fragment (16x32, bf16 LDS): v16bf = row[kbase+g*16..+16)
__device__ __forceinline__ v16bf frag_a(const bf16_t* base, int mbase, int kbase, int lane) {
  const int g = lane >> 4, m = lane & 15;
  const bf16_t* p = base + (mbase + m) * LDK + kbase + (g << 4);
  FragU u;
  u.h[0] = *reinterpret_cast<const v8bf*>(p);
  u.h[1] = *reinterpret_cast<const v8bf*>(p + 8);
  return u.v;
}

// B fragment (32x16) from a bf16 [n][k] LDS tile: v16bf = row[kbase+g*16..+16)
__device__ __forceinline__ v16bf frag_b(const bf16_t* base, int nbase, int kbase, int lane) {
  const int g = lane >> 4, n = lane & 15;
  const bf16_t* p = base + (nbase + n) * LDK + kbase + (g << 4);
  FragU u;
  u.h[0] = *reinterpret_cast<const v8bf*>(p);
  u.h[1] = *reinterpret_cast<const v8bf*>(p + 8);
  return u.v;
}

// 8 contiguous f32 -> v8bf (2x ds_load_b128 + 4x v_cvt_pk_bf16_f32)
__device__ __forceinline__ v8bf cvt8(const float* p) {
  v8bf r;
#pragma unroll
  for (int i = 0; i < 8; ++i) r[i] = (bf16_t)p[i];
  return r;
}

// B fragment (32x16) from an f32 [n][k] LDS tile (async-DMA'd weights).
__device__ __forceinline__ v16bf frag_b_f32(const float* base, int nbase, int kbase, int lane) {
  const int g = lane >> 4, n = lane & 15;
  const float* p = base + (nbase + n) * LDKF + kbase + (g << 4);
  FragU u;
  u.h[0] = cvt8(p);
  u.h[1] = cvt8(p + 8);
  return u.v;
}

__device__ __forceinline__ v8f wmma_bf16(v16bf a, v16bf b, v8f c) {
  return __builtin_amdgcn_wmma_f32_16x16x32_bf16(false, a, false, b, (short)0, c, false, false);
}

__device__ __forceinline__ v8f v8f_zero() {
  v8f z = {0.f, 0.f, 0.f, 0.f, 0.f, 0.f, 0.f, 0.f};
  return z;
}

// float4 -> 4x bf16 packed 64-bit LDS store.
__device__ __forceinline__ void st4(bf16_t* dst, float4 f) {
  v4bf p;
  p[0] = (bf16_t)f.x; p[1] = (bf16_t)f.y; p[2] = (bf16_t)f.z; p[3] = (bf16_t)f.w;
  *reinterpret_cast<v4bf*>(dst) = p;
}
__device__ __forceinline__ float4 ld4(const float* p) {
  return *reinterpret_cast<const float4*>(p);
}

// Async DMA: 16B per lane, global -> LDS, tracked by ASYNCcnt (ISA 15.18.3
// op 98).  VDST = LDS byte address (low 32 bits of the generic pointer),
// VADDR = 64-bit global address, saddr = off.
__device__ __forceinline__ void async_copy_b128(const float* g, float* lds) {
  const unsigned l = (unsigned)(uintptr_t)lds;
  asm volatile("global_load_async_to_lds_b128 %0, %1, off"
               :: "v"(l), "v"(g) : "memory");
}
__device__ __forceinline__ void wait_asynccnt0() {
  asm volatile("s_wait_asynccnt 0x0" ::: "memory");
}

// ---------------------------------------------------------------------------
// Generic GEMM: C[M,N] = act(A[M,K] @ W[N,K]^T + bias) (+ res).
// Block 256 threads (8 waves), tile 64(M) x 128(N), K-tile 64.
// W tile: async DMA to f32 LDS (no VGPR staging, overlaps A staging).
// A tile: float4 loads + packed bf16 LDS stores.
// ACT: 0=none, 1=exact GELU.  RES: add res[row*ldc+col].
// ---------------------------------------------------------------------------
template <int ACT, int RES, int K>
__global__ __launch_bounds__(256) void gemm_kernel(
    const float* __restrict__ A, int lda,
    const float* __restrict__ W,       // [N,K] row-major (torch weight)
    const float* __restrict__ bias,    // [N]
    const float* __restrict__ res,     // [M,ldc] when RES
    float* __restrict__ C, int ldc) {
  __shared__ bf16_t As[64][LDK];                    // activations, bf16
  __shared__ __align__(16) float Bf[128][LDKF];     // weights, f32 (async DMA)
  const int tid = threadIdx.x;
  const int lane = tid & 31, wave = tid >> 5;
  const int tm = blockIdx.y * 64, tn = blockIdx.x * 128;
  const int wm = (wave & 1) * 32, wn = (wave >> 1) * 32;

  v8f acc[2][2];
  acc[0][0] = v8f_zero(); acc[0][1] = v8f_zero();
  acc[1][0] = v8f_zero(); acc[1][1] = v8f_zero();

  for (int k0 = 0; k0 < K; k0 += 64) {
    // 1) kick off async DMA of the 128x64 f32 weight tile (8 instr/wave).
#pragma unroll
    for (int i = 0; i < 8; ++i) {
      const int e = tid + (i << 8);
      const int n = e >> 4, c4 = (e & 15) << 2;
      async_copy_b128(&W[(size_t)(tn + n) * K + k0 + c4], &Bf[n][c4]);
    }
    // 2) stage the 64x64 activation tile through VGPRs with bf16 convert
    //    (overlaps with the in-flight async DMA above).
#pragma unroll
    for (int i = 0; i < 4; ++i) {
      const int e = tid + (i << 8);
      const int r = e >> 4, c4 = (e & 15) << 2;
      st4(&As[r][c4], ld4(&A[(size_t)(tm + r) * lda + k0 + c4]));
    }
    wait_asynccnt0();
    __syncthreads();

#pragma unroll
    for (int kk = 0; kk < 64; kk += 32) {
      const v16bf a0 = frag_a(&As[0][0], wm,      kk, lane);
      const v16bf a1 = frag_a(&As[0][0], wm + 16, kk, lane);
      const v16bf b0 = frag_b_f32(&Bf[0][0], wn,      kk, lane);
      const v16bf b1 = frag_b_f32(&Bf[0][0], wn + 16, kk, lane);
      acc[0][0] = wmma_bf16(a0, b0, acc[0][0]);
      acc[0][1] = wmma_bf16(a0, b1, acc[0][1]);
      acc[1][0] = wmma_bf16(a1, b0, acc[1][0]);
      acc[1][1] = wmma_bf16(a1, b1, acc[1][1]);
    }
    __syncthreads();
  }

  const int g = lane >> 4, nl = lane & 15;
#pragma unroll
  for (int sm = 0; sm < 2; ++sm)
#pragma unroll
    for (int sn = 0; sn < 2; ++sn)
#pragma unroll
      for (int j = 0; j < 8; ++j) {
        const int row = tm + wm + sm * 16 + (g << 3) + j;
        const int col = tn + wn + sn * 16 + nl;
        float v = acc[sm][sn][j] + bias[col];
        if (ACT == 1) v = 0.5f * v * (1.0f + erff(v * 0.70710678118654752f));
        if (RES) v += res[(size_t)row * ldc + col];
        C[(size_t)row * ldc + col] = v;
      }
}

// ---------------------------------------------------------------------------
// G[bh, l, p] = sum_d Q[b,h,l,d] * rel_table[p, h*64+d],  p in [0, 2S-1).
// K = HD = 64 -> single LDS pass, straight-line 8 WMMA.  Grid (4, 4, B*H).
// ---------------------------------------------------------------------------
__global__ __launch_bounds__(256) void gemm_rel_kernel(
    const float* __restrict__ qkv, const float* __restrict__ rel_table,
    float* __restrict__ G) {
  __shared__ bf16_t As[64][LDK];
  __shared__ bf16_t Bt[128][LDK];
  const int tid = threadIdx.x, lane = tid & 31, wave = tid >> 5;
  const int bh = blockIdx.z, b = bh >> 3, h = bh & 7;
  const int tm = blockIdx.y * 64, tn = blockIdx.x * 128;
  const int wm = (wave & 1) * 32, wn = (wave >> 1) * 32;
  const float* Aq = qkv + (size_t)b * S_ * (3 * E_) + h * HD_;

#pragma unroll
  for (int i = 0; i < 4; ++i) {
    const int e = tid + (i << 8);
    const int r = e >> 4, c4 = (e & 15) << 2;
    st4(&As[r][c4], ld4(&Aq[(size_t)(tm + r) * (3 * E_) + c4]));
  }
  const float4 f0 = {0.f, 0.f, 0.f, 0.f};
#pragma unroll
  for (int i = 0; i < 8; ++i) {
    const int e = tid + (i << 8);
    const int n = e >> 4, c4 = (e & 15) << 2;
    const int p = tn + n;
    st4(&Bt[n][c4], (p < 2 * S_ - 1)
                        ? ld4(&rel_table[(size_t)p * E_ + h * HD_ + c4])
                        : f0);
  }
  __syncthreads();

  v8f acc[2][2];
  acc[0][0] = v8f_zero(); acc[0][1] = v8f_zero();
  acc[1][0] = v8f_zero(); acc[1][1] = v8f_zero();
#pragma unroll
  for (int kk = 0; kk < 64; kk += 32) {
    const v16bf a0 = frag_a(&As[0][0], wm,      kk, lane);
    const v16bf a1 = frag_a(&As[0][0], wm + 16, kk, lane);
    const v16bf b0 = frag_b(&Bt[0][0], wn,      kk, lane);
    const v16bf b1 = frag_b(&Bt[0][0], wn + 16, kk, lane);
    acc[0][0] = wmma_bf16(a0, b0, acc[0][0]);
    acc[0][1] = wmma_bf16(a0, b1, acc[0][1]);
    acc[1][0] = wmma_bf16(a1, b0, acc[1][0]);
    acc[1][1] = wmma_bf16(a1, b1, acc[1][1]);
  }

  const int g = lane >> 4, nl = lane & 15;
#pragma unroll
  for (int sm = 0; sm < 2; ++sm)
#pragma unroll
    for (int sn = 0; sn < 2; ++sn)
#pragma unroll
      for (int j = 0; j < 8; ++j) {
        const int row = tm + wm + sm * 16 + (g << 3) + j;
        const int col = tn + wn + sn * 16 + nl;
        G[((size_t)bh * S_ + row) * 512 + col] = acc[sm][sn][j];
      }
}

// ---------------------------------------------------------------------------
// scores[bh,l,m] = (1/8)*Q.K^T + G[bh,l,m-l+255] + 1.0*(m>l)
// (the reference adds the *bool* triu mask as +1.0, not -inf).  Grid (2,4,B*H).
// ---------------------------------------------------------------------------
__global__ __launch_bounds__(256) void gemm_qk_kernel(
    const float* __restrict__ qkv, const float* __restrict__ G,
    float* __restrict__ scores) {
  __shared__ bf16_t As[64][LDK];
  __shared__ bf16_t Bt[128][LDK];
  const int tid = threadIdx.x, lane = tid & 31, wave = tid >> 5;
  const int bh = blockIdx.z, b = bh >> 3, h = bh & 7;
  const int tm = blockIdx.y * 64, tn = blockIdx.x * 128;
  const int wm = (wave & 1) * 32, wn = (wave >> 1) * 32;
  const float* Q  = qkv + (size_t)b * S_ * (3 * E_) + h * HD_;
  const float* Kp = qkv + (size_t)b * S_ * (3 * E_) + E_ + h * HD_;

#pragma unroll
  for (int i = 0; i < 4; ++i) {
    const int e = tid + (i << 8);
    const int r = e >> 4, c4 = (e & 15) << 2;
    st4(&As[r][c4], ld4(&Q[(size_t)(tm + r) * (3 * E_) + c4]));
  }
#pragma unroll
  for (int i = 0; i < 8; ++i) {
    const int e = tid + (i << 8);
    const int n = e >> 4, c4 = (e & 15) << 2;
    st4(&Bt[n][c4], ld4(&Kp[(size_t)(tn + n) * (3 * E_) + c4]));
  }
  __syncthreads();

  v8f acc[2][2];
  acc[0][0] = v8f_zero(); acc[0][1] = v8f_zero();
  acc[1][0] = v8f_zero(); acc[1][1] = v8f_zero();
#pragma unroll
  for (int kk = 0; kk < 64; kk += 32) {
    const v16bf a0 = frag_a(&As[0][0], wm,      kk, lane);
    const v16bf a1 = frag_a(&As[0][0], wm + 16, kk, lane);
    const v16bf b0 = frag_b(&Bt[0][0], wn,      kk, lane);
    const v16bf b1 = frag_b(&Bt[0][0], wn + 16, kk, lane);
    acc[0][0] = wmma_bf16(a0, b0, acc[0][0]);
    acc[0][1] = wmma_bf16(a0, b1, acc[0][1]);
    acc[1][0] = wmma_bf16(a1, b0, acc[1][0]);
    acc[1][1] = wmma_bf16(a1, b1, acc[1][1]);
  }

  const int g = lane >> 4, nl = lane & 15;
#pragma unroll
  for (int sm = 0; sm < 2; ++sm)
#pragma unroll
    for (int sn = 0; sn < 2; ++sn)
#pragma unroll
      for (int j = 0; j < 8; ++j) {
        const int l = tm + wm + sm * 16 + (g << 3) + j;
        const int m = tn + wn + sn * 16 + nl;
        const float v = acc[sm][sn][j] * 0.125f
                      + G[((size_t)bh * S_ + l) * 512 + (m - l + (S_ - 1))]
                      + ((m > l) ? 1.0f : 0.0f);
        scores[((size_t)bh * S_ + l) * S_ + m] = v;
      }
}

// ---------------------------------------------------------------------------
// Row-wise softmax over S=256 (one 256-thread block per row).
// ---------------------------------------------------------------------------
__global__ __launch_bounds__(256) void softmax_kernel(float* __restrict__ sc) {
  const int row = blockIdx.x, tid = threadIdx.x;
  __shared__ float red[256];
  float v = sc[(size_t)row * S_ + tid];
  red[tid] = v;
  __syncthreads();
  for (int s = 128; s; s >>= 1) {
    if (tid < s) red[tid] = fmaxf(red[tid], red[tid + s]);
    __syncthreads();
  }
  const float mx = red[0];
  __syncthreads();
  const float e = __expf(v - mx);
  red[tid] = e;
  __syncthreads();
  for (int s = 128; s; s >>= 1) {
    if (tid < s) red[tid] += red[tid + s];
    __syncthreads();
  }
  sc[(size_t)row * S_ + tid] = e / red[0];
}

// ---------------------------------------------------------------------------
// o[b,l,h*64+d] = sum_m attn[bh,l,m] * V[b,m,h*64+d].  M=256,N=64,K=256.
// 128 threads (4 waves), tile 64x64, K-tile 64.  Grid (1, 4, B*H).
// ---------------------------------------------------------------------------
__global__ __launch_bounds__(128) void gemm_av_kernel(
    const float* __restrict__ sc, const float* __restrict__ qkv,
    float* __restrict__ o) {
  __shared__ bf16_t As[64][LDK];
  __shared__ bf16_t Bt[64][LDK];
  const int tid = threadIdx.x, lane = tid & 31, wave = tid >> 5;
  const int bh = blockIdx.z, b = bh >> 3, h = bh & 7;
  const int tm = blockIdx.y * 64;
  const int wm = (wave & 1) * 32, wn = (wave >> 1) * 32;
  const float* Vp = qkv + (size_t)b * S_ * (3 * E_) + 2 * E_ + h * HD_;

  v8f acc[2][2];
  acc[0][0] = v8f_zero(); acc[0][1] = v8f_zero();
  acc[1][0] = v8f_zero(); acc[1][1] = v8f_zero();

  for (int k0 = 0; k0 < S_; k0 += 64) {
#pragma unroll
    for (int i = 0; i < 8; ++i) {            // attn tile 64x64 = 1024 float4
      const int e = tid + (i << 7);
      const int r = e >> 4, c4 = (e & 15) << 2;
      st4(&As[r][c4], ld4(&sc[((size_t)bh * S_ + tm + r) * S_ + k0 + c4]));
    }
#pragma unroll
    for (int i = 0; i < 8; ++i) {            // V tile 64(k) x 64(n), transpose
      const int e = tid + (i << 7);
      const int k = e >> 4, n4 = (e & 15) << 2;
      const float4 f = ld4(&Vp[(size_t)(k0 + k) * (3 * E_) + n4]);
      Bt[n4 + 0][k] = (bf16_t)f.x;
      Bt[n4 + 1][k] = (bf16_t)f.y;
      Bt[n4 + 2][k] = (bf16_t)f.z;
      Bt[n4 + 3][k] = (bf16_t)f.w;
    }
    __syncthreads();
#pragma unroll
    for (int kk = 0; kk < 64; kk += 32) {
      const v16bf a0 = frag_a(&As[0][0], wm,      kk, lane);
      const v16bf a1 = frag_a(&As[0][0], wm + 16, kk, lane);
      const v16bf b0 = frag_b(&Bt[0][0], wn,      kk, lane);
      const v16bf b1 = frag_b(&Bt[0][0], wn + 16, kk, lane);
      acc[0][0] = wmma_bf16(a0, b0, acc[0][0]);
      acc[0][1] = wmma_bf16(a0, b1, acc[0][1]);
      acc[1][0] = wmma_bf16(a1, b0, acc[1][0]);
      acc[1][1] = wmma_bf16(a1, b1, acc[1][1]);
    }
    __syncthreads();
  }

  const int g = lane >> 4, nl = lane & 15;
#pragma unroll
  for (int sm = 0; sm < 2; ++sm)
#pragma unroll
    for (int sn = 0; sn < 2; ++sn)
#pragma unroll
      for (int j = 0; j < 8; ++j) {
        const int row = tm + wm + sm * 16 + (g << 3) + j;
        const int col = wn + sn * 16 + nl;
        o[((size_t)b * S_ + row) * E_ + h * HD_ + col] = acc[sm][sn][j];
      }
}

// ---------------------------------------------------------------------------
// LayerNorm over E=512 (one 256-thread block per row, 2 elements/thread).
// ---------------------------------------------------------------------------
__global__ __launch_bounds__(256) void layernorm_kernel(
    const float* __restrict__ in, const float* __restrict__ s,
    const float* __restrict__ bsh, float* __restrict__ out) {
  const int row = blockIdx.x, tid = threadIdx.x;
  __shared__ float r1[256], r2[256];
  const float a = in[(size_t)row * E_ + tid];
  const float c = in[(size_t)row * E_ + tid + 256];
  r1[tid] = a + c;
  r2[tid] = a * a + c * c;
  __syncthreads();
  for (int st = 128; st; st >>= 1) {
    if (tid < st) { r1[tid] += r1[tid + st]; r2[tid] += r2[tid + st]; }
    __syncthreads();
  }
  const float mean = r1[0] * (1.0f / E_);
  const float var  = r2[0] * (1.0f / E_) - mean * mean;
  const float inv  = rsqrtf(var + 1e-5f);
  out[(size_t)row * E_ + tid]       = (a - mean) * inv * s[tid] + bsh[tid];
  out[(size_t)row * E_ + tid + 256] = (c - mean) * inv * s[tid + 256] + bsh[tid + 256];
}

// Embedding gather * sqrt(E), then input LayerNorm.
__global__ __launch_bounds__(256) void embed_ln_kernel(
    const int* __restrict__ src, const float* __restrict__ emb,
    const float* __restrict__ s, const float* __restrict__ bsh,
    float* __restrict__ out) {
  const int row = blockIdx.x, tid = threadIdx.x;
  __shared__ float r1[256], r2[256];
  const float* er = emb + (size_t)src[row] * E_;
  const float a = er[tid] * SQRT_E;
  const float c = er[tid + 256] * SQRT_E;
  r1[tid] = a + c;
  r2[tid] = a * a + c * c;
  __syncthreads();
  for (int st = 128; st; st >>= 1) {
    if (tid < st) { r1[tid] += r1[tid + st]; r2[tid] += r2[tid + st]; }
    __syncthreads();
  }
  const float mean = r1[0] * (1.0f / E_);
  const float var  = r2[0] * (1.0f / E_) - mean * mean;
  const float inv  = rsqrtf(var + 1e-5f);
  out[(size_t)row * E_ + tid]       = (a - mean) * inv * s[tid] + bsh[tid];
  out[(size_t)row * E_ + tid + 256] = (c - mean) * inv * s[tid + 256] + bsh[tid + 256];
}

// ---------------------------------------------------------------------------
extern "C" void kernel_launch(void* const* d_in, const int* in_sizes, int n_in,
                              void* d_out, int out_size, void* d_ws, size_t ws_size,
                              hipStream_t stream) {
  (void)in_sizes; (void)n_in; (void)out_size; (void)ws_size;
  const int*   src   = (const int*)d_in[0];
  const float* emb   = (const float*)d_in[1];
  const float* rel   = (const float*)d_in[2];
  const float* nin_s = (const float*)d_in[3];
  const float* nin_b = (const float*)d_in[4];
  const float* inW   = (const float*)d_in[5];
  const float* inB   = (const float*)d_in[6];
  const float* outW  = (const float*)d_in[7];
  const float* outB  = (const float*)d_in[8];
  const float* ln1s  = (const float*)d_in[9];
  const float* ln1b  = (const float*)d_in[10];
  const float* ln2s  = (const float*)d_in[11];
  const float* ln2b  = (const float*)d_in[12];
  const float* w1    = (const float*)d_in[13];
  const float* b1    = (const float*)d_in[14];
  const float* w2    = (const float*)d_in[15];
  const float* b2    = (const float*)d_in[16];
  const float* nf_s  = (const float*)d_in[17];
  const float* nf_b  = (const float*)d_in[18];
  const float* dec_w = (const float*)d_in[19];
  const float* dec_b = (const float*)d_in[20];

  float* ws  = (float*)d_ws;                    // ~46.1 MB f32 total
  float* x   = ws;                              //  524288  [M,E]
  float* h   = x   + (size_t)M_ * E_;           //  524288  [M,E]
  float* qkv = h   + (size_t)M_ * E_;           // 1572864  [M,3E]
  float* o   = qkv + (size_t)M_ * 3 * E_;       //  524288  [M,E]
  float* ff  = o   + (size_t)M_ * E_;           // 2097152  [M,F]
  float* G   = ff  + (size_t)M_ * F_;           // 4194304  [B*H,S,512]
  float* sc  = G   + (size_t)B_ * H_ * S_ * 512;// 2097152  [B*H,S,S]

  embed_ln_kernel<<<M_, 256, 0, stream>>>(src, emb, nin_s, nin_b, x);

  for (int l = 0; l < L_; ++l) {
    layernorm_kernel<<<M_, 256, 0, stream>>>(x, ln1s + l * E_, ln1b + l * E_, h);
    gemm_kernel<0, 0, E_><<<dim3(12, 16), 256, 0, stream>>>(
        h, E_, inW + (size_t)l * 3 * E_ * E_, inB + l * 3 * E_, nullptr,
        qkv, 3 * E_);
    gemm_rel_kernel<<<dim3(4, 4, B_ * H_), 256, 0, stream>>>(qkv, rel, G);
    gemm_qk_kernel<<<dim3(2, 4, B_ * H_), 256, 0, stream>>>(qkv, G, sc);
    softmax_kernel<<<B_ * H_ * S_, 256, 0, stream>>>(sc);
    gemm_av_kernel<<<dim3(1, 4, B_ * H_), 128, 0, stream>>>(sc, qkv, o);
    gemm_kernel<0, 1, E_><<<dim3(4, 16), 256, 0, stream>>>(
        o, E_, outW + (size_t)l * E_ * E_, outB + l * E_, x, x, E_);
    layernorm_kernel<<<M_, 256, 0, stream>>>(x, ln2s + l * E_, ln2b + l * E_, h);
    gemm_kernel<1, 0, E_><<<dim3(16, 16), 256, 0, stream>>>(
        h, E_, w1 + (size_t)l * F_ * E_, b1 + l * F_, nullptr, ff, F_);
    gemm_kernel<0, 1, F_><<<dim3(4, 16), 256, 0, stream>>>(
        ff, F_, w2 + (size_t)l * E_ * F_, b2 + l * E_, x, x, E_);
  }

  layernorm_kernel<<<M_, 256, 0, stream>>>(x, nf_s, nf_b, h);
  gemm_kernel<0, 0, E_><<<dim3(250, 16), 256, 0, stream>>>(
      h, E_, dec_w, dec_b, nullptr, (float*)d_out, V_);
}
